// MoAETrasnformer_89850715833126
// MI455X (gfx1250) — matrix operand order repaired and checked by reading, not compiled, not run on hardware
//
#include <hip/hip_runtime.h>
#include <hip/hip_bf16.h>
#include <hip/hip_fp16.h>
#include <math.h>

typedef __attribute__((ext_vector_type(16))) _Float16 v16h;
typedef __attribute__((ext_vector_type(8)))  float    v8f;
typedef __attribute__((ext_vector_type(4)))  int      v4i;

// Problem constants (from reference)
constexpr int Vv = 32000, Dd = 1024, Hh = 16, Ll = 2, Ss = 1024, Bb = 2;
constexpr int Ee = 8, TOPN = 2, Ff = 4096;
constexpr int Nn = Bb * Ss;                  // 2048 tokens
constexpr int CAP = 640;                     // int(1.25 * 2 * 2048 / 8)

union AFrag { v16h v; _Float16 h[16]; };
union CFrag { v8f  v; float    f[8];  };

// Detect the gfx1250 async global->LDS builtin; fall back to sync staging.
#if defined(__has_builtin)
#if __has_builtin(__builtin_amdgcn_global_load_async_to_lds_b128)
#define ASYNC_OK 1
#endif
#endif
#ifndef ASYNC_OK
#define ASYNC_OK 0
#endif

#if ASYNC_OK
// Signature (from clang diagnostic): (v4i AS1*, v4i AS3*, imm offset, imm cpol)
typedef __attribute__((address_space(1))) v4i* gv4p;
typedef __attribute__((address_space(3))) v4i* lv4p;
__device__ __forceinline__ void async_copy16(const float* g, float* l) {
  __builtin_amdgcn_global_load_async_to_lds_b128((gv4p)(g), (lv4p)(l), 0, 0);
}
#endif

// ---------------------------------------------------------------------------
// Embedding: x[n,d] = emb[tok[n],d] + pos_emb[n % S, d]
// ---------------------------------------------------------------------------
__global__ void k_embed(const int* __restrict__ tok, const float* __restrict__ emb,
                        const float* __restrict__ pe, float* __restrict__ x) {
  const int n = blockIdx.x;
  const int t = tok[n];
  const int s = n & (Ss - 1);
  for (int d = threadIdx.x; d < Dd; d += blockDim.x)
    x[(size_t)n * Dd + d] = emb[(size_t)t * Dd + d] + pe[(size_t)s * Dd + d];
}

// ---------------------------------------------------------------------------
// WMMA GEMM: C = A(MxK,f32) @ B(KxN,f32) [+bias] [gelu], f16 compute f32 acc.
// Block tile 128x128, 8 waves (4 along M x 2 along N), each wave 32x64.
// flags: bit0 = add bias[n], bit1 = tanh-GELU epilogue.
// Batched over blockIdx.z with element strides sA/sB/sC.
// Requires M%128==0, N%128==0, K%32==0 (true for every call site here).
//
// ASYNC path: fp32 tiles double-buffered in LDS via global_load_async_to_lds,
// one s_wait_asynccnt + one barrier per K-step; next stage issued right after
// the barrier so the async engine overlaps the 8 WMMAs. A-tile chunks are
// XOR-swizzled across banks (free: async lets us choose each chunk's LDS
// address). fp32->fp16 conversion happens at fragment gather (co-executes
// with the XDL WMMA pipe).
// ---------------------------------------------------------------------------
__global__ __launch_bounds__(256) void k_gemm(
    const float* __restrict__ A, const float* __restrict__ B,
    float* __restrict__ C, const float* __restrict__ bias,
    int M, int N, int K, long long sA, long long sB, long long sC, int flags) {
  const int tid  = threadIdx.x;
  const int lane = tid & 31;
  const int wave = tid >> 5;
  const int wm   = wave & 3;                    // wave row (0..3)  -> 32 rows
  const int wn   = wave >> 2;                   // wave col (0..1)  -> 64 cols
  const int bm   = blockIdx.y * 128;
  const int bn   = blockIdx.x * 128;
  A += (long long)blockIdx.z * sA;
  B += (long long)blockIdx.z * sB;
  C += (long long)blockIdx.z * sC;

  v8f acc[2][4];
#pragma unroll
  for (int mi = 0; mi < 2; ++mi)
#pragma unroll
    for (int ni = 0; ni < 4; ++ni)
      acc[mi][ni] = (v8f){0.f, 0.f, 0.f, 0.f, 0.f, 0.f, 0.f, 0.f};

  const int l15 = lane & 15;
  const int hi  = lane >> 4;

#if ASYNC_OK
  // ---------------- asynchronous double-buffered staging ----------------
  __shared__ __align__(16) float Abuf[2][128 * 32];   // [row][k] XOR-swizzled
  __shared__ __align__(16) float Bbuf[2][32 * 128];   // [k][n]

  auto issue = [&](int st, int k0) {
    // A tile: 1024 16B-chunks; 4 per thread. chunk = (row, cc) cc in 0..7
#pragma unroll
    for (int j = 0; j < 4; ++j) {
      const int chunk = j * 256 + tid;
      const int row = chunk >> 3, cc = chunk & 7;
      const float* g = &A[(long long)(bm + row) * K + k0 + cc * 4];
      float* l = &Abuf[st][row * 32 + ((cc ^ (row & 7)) << 2)];
      async_copy16(g, l);
    }
    // B tile: chunk = (k, cc) cc in 0..31
#pragma unroll
    for (int j = 0; j < 4; ++j) {
      const int chunk = j * 256 + tid;
      const int k = chunk >> 5, cc = chunk & 31;
      const float* g = &B[(long long)(k0 + k) * N + bn + cc * 4];
      float* l = &Bbuf[st][k * 128 + cc * 4];
      async_copy16(g, l);
    }
  };

  const int nIter = K >> 5;
  issue(0, 0);
  for (int it = 0; it < nIter; ++it) {
    asm volatile("s_wait_asynccnt 0x0" ::: "memory");  // own stage landed
    __syncthreads();                                   // all waves landed; prev bufs free
    if (it + 1 < nIter) issue((it + 1) & 1, (it + 1) * 32);
    const int st = it & 1;

    AFrag af[2], bf[4];
    const int s7 = l15 & 7;                            // (row & 7) for swizzle
#pragma unroll
    for (int mi = 0; mi < 2; ++mi) {
      const float* pA = &Abuf[st][(wm * 32 + mi * 16 + l15) * 32];
#pragma unroll
      for (int v = 0; v < 8; ++v) {
        const int k  = (v < 4) ? (hi * 8 + 2 * v) : (16 + hi * 8 + 2 * (v - 4));
        const int ix = (((k >> 2) ^ s7) << 2) | (k & 3);
        af[mi].h[2 * v]     = (_Float16)pA[ix];
        af[mi].h[2 * v + 1] = (_Float16)pA[ix + 1];
      }
    }
#pragma unroll
    for (int ni = 0; ni < 4; ++ni) {
      const float* pB = &Bbuf[st][wn * 64 + ni * 16 + l15];
#pragma unroll
      for (int v = 0; v < 8; ++v) {
        const int k = (v < 4) ? (hi * 8 + 2 * v) : (16 + hi * 8 + 2 * (v - 4));
        bf[ni].h[2 * v]     = (_Float16)pB[(size_t)k * 128];
        bf[ni].h[2 * v + 1] = (_Float16)pB[(size_t)(k + 1) * 128];
      }
    }

#pragma unroll
    for (int mi = 0; mi < 2; ++mi)
#pragma unroll
      for (int ni = 0; ni < 4; ++ni)
        acc[mi][ni] = __builtin_amdgcn_wmma_f32_16x16x32_f16(
            false, af[mi].v, false, bf[ni].v, (short)0, acc[mi][ni], false, false);
  }
#else
  // ---------------- synchronous staging fallback ----------------
  constexpr int LDP = 34;
  __shared__ _Float16 As[128 * LDP];            // A tile [row][k]
  __shared__ _Float16 Bs[128 * LDP];            // B tile transposed [col][k]

  for (int k0 = 0; k0 < K; k0 += 32) {
    {
      const int row = tid >> 3;
      const int col = (tid & 7) * 4;
#pragma unroll
      for (int i = 0; i < 4; ++i) {
        const float4 v = *(const float4*)&A[(long long)(bm + row + i * 32) * K + k0 + col];
        _Float16* d = &As[(row + i * 32) * LDP + col];
        d[0] = (_Float16)v.x; d[1] = (_Float16)v.y;
        d[2] = (_Float16)v.z; d[3] = (_Float16)v.w;
      }
      const int kr = tid >> 5;
      const int n0 = (tid & 31) * 4;
#pragma unroll
      for (int i = 0; i < 4; ++i) {
        const float4 v = *(const float4*)&B[(long long)(k0 + kr + i * 8) * N + bn + n0];
        const int kk = kr + i * 8;
        Bs[(n0 + 0) * LDP + kk] = (_Float16)v.x;
        Bs[(n0 + 1) * LDP + kk] = (_Float16)v.y;
        Bs[(n0 + 2) * LDP + kk] = (_Float16)v.z;
        Bs[(n0 + 3) * LDP + kk] = (_Float16)v.w;
      }
    }
    __syncthreads();

    AFrag af[2], bf[4];
#pragma unroll
    for (int mi = 0; mi < 2; ++mi) {
      const _Float16* p = &As[(wm * 32 + mi * 16 + l15) * LDP];
#pragma unroll
      for (int v = 0; v < 4; ++v) {
        const int k = hi * 8 + 2 * v;
        af[mi].h[2 * v] = p[k]; af[mi].h[2 * v + 1] = p[k + 1];
      }
#pragma unroll
      for (int v = 0; v < 4; ++v) {
        const int k = 16 + hi * 8 + 2 * v;
        af[mi].h[8 + 2 * v] = p[k]; af[mi].h[8 + 2 * v + 1] = p[k + 1];
      }
    }
#pragma unroll
    for (int ni = 0; ni < 4; ++ni) {
      const _Float16* p = &Bs[(wn * 64 + ni * 16 + l15) * LDP];
#pragma unroll
      for (int v = 0; v < 8; ++v) {
        const int k = (v < 4) ? (hi * 8 + 2 * v) : (16 + hi * 8 + 2 * (v - 4));
        bf[ni].h[2 * v] = p[k]; bf[ni].h[2 * v + 1] = p[k + 1];
      }
    }

#pragma unroll
    for (int mi = 0; mi < 2; ++mi)
#pragma unroll
      for (int ni = 0; ni < 4; ++ni)
        acc[mi][ni] = __builtin_amdgcn_wmma_f32_16x16x32_f16(
            false, af[mi].v, false, bf[ni].v, (short)0, acc[mi][ni], false, false);
    __syncthreads();
  }
#endif

  // ---- epilogue: D-matrix layout  m = 8*hi + j (vgpr), n = lane&15
#pragma unroll
  for (int mi = 0; mi < 2; ++mi)
#pragma unroll
    for (int ni = 0; ni < 4; ++ni) {
      CFrag u; u.v = acc[mi][ni];
      const int n = bn + wn * 64 + ni * 16 + l15;
      const float bv = (flags & 1) ? bias[n] : 0.f;
#pragma unroll
      for (int j = 0; j < 8; ++j) {
        const int m = bm + wm * 32 + mi * 16 + hi * 8 + j;
        float v = u.f[j] + bv;
        if (flags & 2) {                        // tanh-approx GELU (jax default)
          const float t = 0.7978845608028654f * (v + 0.044715f * v * v * v);
          v = 0.5f * v * (1.0f + tanhf(t));
        }
        C[(long long)m * N + n] = v;
      }
    }
}

// ---------------------------------------------------------------------------
// Attention: one wave per (b,h,q) row, online softmax over keys t<=q.
// dh = 64 -> each lane owns 2 head dims. qkv layout [n][3D], q|k|v slabs.
// ---------------------------------------------------------------------------
__global__ __launch_bounds__(256) void k_attn(const float* __restrict__ qkv,
                                              float* __restrict__ out) {
  const int lane = threadIdx.x & 31;
  const int rid  = blockIdx.x * 8 + (threadIdx.x >> 5);
  const int q    = rid & (Ss - 1);
  const int h    = (rid >> 10) & (Hh - 1);
  const int b    = rid >> 14;
  const int d0   = lane * 2;
  const float scale = 0.125f;                   // 1/sqrt(64)

  const float* qr = qkv + (size_t)(b * Ss + q) * 3 * Dd + h * 64;
  const float q0 = qr[d0] * scale, q1 = qr[d0 + 1] * scale;

  float m = -INFINITY, l = 0.f, a0 = 0.f, a1 = 0.f;
  for (int t = 0; t <= q; ++t) {
    const float* kr = qkv + (size_t)(b * Ss + t) * 3 * Dd + Dd + h * 64;
    float part = q0 * kr[d0] + q1 * kr[d0 + 1];
#pragma unroll
    for (int off = 16; off > 0; off >>= 1) part += __shfl_xor(part, off, 32);
    const float mn = fmaxf(m, part);
    const float c  = __expf(m - mn);
    const float p  = __expf(part - mn);
    const float* vr = kr + Dd;
    l  = l * c + p;
    a0 = a0 * c + p * vr[d0];
    a1 = a1 * c + p * vr[d0 + 1];
    m  = mn;
  }
  float* o = out + (size_t)(b * Ss + q) * Dd + h * 64;
  o[d0] = a0 / l; o[d0 + 1] = a1 / l;
}

// ---------------------------------------------------------------------------
// Fused residual + LayerNorm: x = LN(x + a) * g + b, one block per token.
// ---------------------------------------------------------------------------
__global__ __launch_bounds__(256) void k_add_ln(float* __restrict__ x, const float* __restrict__ a,
                                                const float* __restrict__ g, const float* __restrict__ b) {
  __shared__ float sr[Dd];
  __shared__ float red[256];
  const int n = blockIdx.x, tid = threadIdx.x;
  float local = 0.f;
  for (int d = tid; d < Dd; d += 256) {
    const float r = x[(size_t)n * Dd + d] + a[(size_t)n * Dd + d];
    sr[d] = r; local += r;
  }
  red[tid] = local; __syncthreads();
  for (int s = 128; s > 0; s >>= 1) { if (tid < s) red[tid] += red[tid + s]; __syncthreads(); }
  const float mean = red[0] * (1.0f / Dd);
  __syncthreads();
  local = 0.f;
  for (int d = tid; d < Dd; d += 256) { const float r = sr[d] - mean; local += r * r; }
  red[tid] = local; __syncthreads();
  for (int s = 128; s > 0; s >>= 1) { if (tid < s) red[tid] += red[tid + s]; __syncthreads(); }
  const float inv = rsqrtf(red[0] * (1.0f / Dd) + 1e-5f);
  for (int d = tid; d < Dd; d += 256)
    x[(size_t)n * Dd + d] = (sr[d] - mean) * inv * g[d] + b[d];
}

// ---------------------------------------------------------------------------
// Gate: logits = x[n]@gate_w (D x 8), softmax, top-2; accumulate me / z-loss.
// stats: [0..7]=me_sum, [8..15]=counts, [16]=z_sum, [17]=aux_total
// ---------------------------------------------------------------------------
__global__ __launch_bounds__(256) void k_gate(const float* __restrict__ x, const float* __restrict__ gw,
                                              float* __restrict__ vals, int* __restrict__ idx,
                                              float* __restrict__ stats) {
  __shared__ float red[256];
  __shared__ float lg[Ee];
  const int n = blockIdx.x, tid = threadIdx.x;
  float acc[Ee];
#pragma unroll
  for (int e = 0; e < Ee; ++e) acc[e] = 0.f;
  for (int d = tid; d < Dd; d += 256) {
    const float xv = x[(size_t)n * Dd + d];
#pragma unroll
    for (int e = 0; e < Ee; ++e) acc[e] += xv * gw[d * Ee + e];
  }
  for (int e = 0; e < Ee; ++e) {
    red[tid] = acc[e]; __syncthreads();
    for (int s = 128; s > 0; s >>= 1) { if (tid < s) red[tid] += red[tid + s]; __syncthreads(); }
    if (tid == 0) lg[e] = red[0];
    __syncthreads();
  }
  if (tid == 0) {
    float mx = -1e30f;
    for (int e = 0; e < Ee; ++e) mx = fmaxf(mx, lg[e]);
    float p[Ee], se = 0.f;
    for (int e = 0; e < Ee; ++e) { p[e] = __expf(lg[e] - mx); se += p[e]; }
    const float lse = mx + __logf(se);
    for (int e = 0; e < Ee; ++e) p[e] /= se;
    int i0 = 0;
    for (int e = 1; e < Ee; ++e) if (p[e] > p[i0]) i0 = e;
    int i1 = -1;
    for (int e = 0; e < Ee; ++e) if (e != i0 && (i1 < 0 || p[e] > p[i1])) i1 = e;
    vals[n * 2] = p[i0]; vals[n * 2 + 1] = p[i1];
    idx[n * 2]  = i0;    idx[n * 2 + 1]  = i1;
    for (int e = 0; e < Ee; ++e) atomicAdd(&stats[e], p[e]);
    atomicAdd(&stats[16], lse * lse);
  }
}

// Deterministic capacity positions: exact reference cumsum semantics
// (full kk=0 pass counts offset the kk=1 pass).
__global__ void k_route(const int* __restrict__ idx, int* __restrict__ pos,
                        float* __restrict__ stats) {
  const int e = threadIdx.x;
  if (e >= Ee) return;
  int c = 0;
  for (int n = 0; n < Nn; ++n) if (idx[n * 2] == e)     pos[n * 2]     = c++;
  for (int n = 0; n < Nn; ++n) if (idx[n * 2 + 1] == e) pos[n * 2 + 1] = c++;
  stats[8 + e] = (float)c;
}

__global__ void k_scatter(const float* __restrict__ x, const int* __restrict__ idx,
                          const int* __restrict__ pos, float* __restrict__ ein) {
  const int a = blockIdx.x, n = a >> 1, kk = a & 1;
  const int p = pos[n * 2 + kk];
  if (p >= CAP) return;
  const int e = idx[n * 2 + kk];
  float* dst = ein + ((size_t)e * CAP + p) * Dd;
  const float* src = x + (size_t)n * Dd;
  for (int d = threadIdx.x; d < Dd; d += blockDim.x) dst[d] = src[d];
}

__global__ void k_combine(const float* __restrict__ y, const float* __restrict__ vals,
                          const int* __restrict__ idx, const int* __restrict__ pos,
                          float* __restrict__ out) {
  const int n = blockIdx.x;
  const int p0 = pos[n * 2], p1 = pos[n * 2 + 1];
  const int e0 = idx[n * 2], e1 = idx[n * 2 + 1];
  const float v0 = (p0 < CAP) ? vals[n * 2] : 0.f;
  const float v1 = (p1 < CAP) ? vals[n * 2 + 1] : 0.f;
  const float* y0 = y + ((size_t)e0 * CAP + (p0 < CAP ? p0 : 0)) * Dd;
  const float* y1 = y + ((size_t)e1 * CAP + (p1 < CAP ? p1 : 0)) * Dd;
  for (int d = threadIdx.x; d < Dd; d += blockDim.x)
    out[(size_t)n * Dd + d] = v0 * y0[d] + v1 * y1[d];
}

// aux += BAL_COEF * balance + Z_COEF * z_loss; reset per-layer stats.
__global__ void k_aux(float* __restrict__ stats) {
  float bal = 0.f;
  for (int e = 0; e < Ee; ++e) {
    const float me = stats[e] / (float)Nn;
    const float ce = stats[8 + e] / (float)(Nn * TOPN);
    bal += me * ce;
  }
  bal *= (float)Ee;
  const float z = stats[16] / (float)Nn;
  stats[17] += 0.01f * bal + 0.001f * z;
  for (int i = 0; i < 17; ++i) stats[i] = 0.f;
}

__global__ void k_writeaux(const float* __restrict__ stats, float* __restrict__ out) {
  out[0] = stats[17];
}

// ---------------------------------------------------------------------------
extern "C" void kernel_launch(void* const* d_in, const int* in_sizes, int n_in,
                              void* d_out, int out_size, void* d_ws, size_t ws_size,
                              hipStream_t stream) {
  const int*   tokens = (const int*)d_in[0];
  const float* emb    = (const float*)d_in[1];
  const float* posemb = (const float*)d_in[2];
  const float* wqkv   = (const float*)d_in[3];
  const float* bqkv   = (const float*)d_in[4];
  const float* wo     = (const float*)d_in[5];
  const float* bo     = (const float*)d_in[6];
  const float* ln1g   = (const float*)d_in[7];
  const float* ln1b   = (const float*)d_in[8];
  const float* ln2g   = (const float*)d_in[9];
  const float* ln2b   = (const float*)d_in[10];
  const float* gatew  = (const float*)d_in[11];
  const float* w1     = (const float*)d_in[12];
  const float* w2     = (const float*)d_in[13];
  const float* headw  = (const float*)d_in[14];
  const float* headb  = (const float*)d_in[15];
  float* out = (float*)d_out;

  float* ws = (float*)d_ws;
  size_t o = 0;
  float* x    = ws + o; o += (size_t)Nn * Dd;
  float* t1   = ws + o; o += (size_t)Nn * Dd;
  float* t2   = ws + o; o += (size_t)Nn * Dd;
  float* qkv  = ws + o; o += (size_t)Nn * 3 * Dd;
  float* vals = ws + o; o += (size_t)Nn * 2;
  float* ein  = ws + o; o += (size_t)Ee * CAP * Dd;
  float* hb   = ws + o; o += (size_t)Ee * CAP * Ff;
  float* yb   = ws + o; o += (size_t)Ee * CAP * Dd;
  float* stats = ws + o; o += 32;
  int* idx  = (int*)(ws + o); o += (size_t)Nn * 2;
  int* posb = (int*)(ws + o); o += (size_t)Nn * 2;

  (void)hipMemsetAsync(stats, 0, 32 * sizeof(float), stream);
  k_embed<<<Nn, 256, 0, stream>>>(tokens, emb, posemb, x);

  for (int l = 0; l < Ll; ++l) {
    // qkv = x @ wqkv[l] + bqkv[l]
    k_gemm<<<dim3(3 * Dd / 128, Nn / 128, 1), 256, 0, stream>>>(
        x, wqkv + (size_t)l * Dd * 3 * Dd, qkv, bqkv + (size_t)l * 3 * Dd,
        Nn, 3 * Dd, Dd, 0, 0, 0, 1);
    // causal multi-head attention
    k_attn<<<(Bb * Hh * Ss) / 8, 256, 0, stream>>>(qkv, t1);
    // out-proj
    k_gemm<<<dim3(Dd / 128, Nn / 128, 1), 256, 0, stream>>>(
        t1, wo + (size_t)l * Dd * Dd, t2, bo + (size_t)l * Dd,
        Nn, Dd, Dd, 0, 0, 0, 1);
    k_add_ln<<<Nn, 256, 0, stream>>>(x, t2, ln1g + l * Dd, ln1b + l * Dd);

    // MoE
    k_gate<<<Nn, 256, 0, stream>>>(x, gatew + (size_t)l * Dd * Ee, vals, idx, stats);
    k_route<<<1, 32, 0, stream>>>(idx, posb, stats);
    (void)hipMemsetAsync(ein, 0, (size_t)Ee * CAP * Dd * sizeof(float), stream);
    k_scatter<<<Nn * 2, 256, 0, stream>>>(x, idx, posb, ein);
    // h = gelu(exp_in @ w1[l]) per expert
    k_gemm<<<dim3(Ff / 128, CAP / 128, Ee), 256, 0, stream>>>(
        ein, w1 + (size_t)l * Ee * Dd * Ff, hb, nullptr,
        CAP, Ff, Dd, (long long)CAP * Dd, (long long)Dd * Ff, (long long)CAP * Ff, 2);
    // y = h @ w2[l] per expert
    k_gemm<<<dim3(Dd / 128, CAP / 128, Ee), 256, 0, stream>>>(
        hb, w2 + (size_t)l * Ee * Ff * Dd, yb, nullptr,
        CAP, Dd, Ff, (long long)CAP * Ff, (long long)Ff * Dd, (long long)CAP * Dd, 0);
    k_combine<<<Nn, 256, 0, stream>>>(yb, vals, idx, posb, t1);
    k_add_ln<<<Nn, 256, 0, stream>>>(x, t1, ln2g + l * Dd, ln2b + l * Dd);
    k_aux<<<1, 1, 0, stream>>>(stats);
  }

  // logits = x @ head_w + head_b  -> d_out[0 .. N*V)
  k_gemm<<<dim3(Vv / 128, Nn / 128, 1), 256, 0, stream>>>(
      x, headw, out, headb, Nn, Vv, Dd, 0, 0, 0, 1);
  k_writeaux<<<1, 1, 0, stream>>>(stats, out + (size_t)Nn * Vv);
}
